// UniMolModel_19877108645985
// MI455X (gfx1250) — matrix operand order repaired and checked by reading, not compiled, not run on hardware
//
#include <hip/hip_runtime.h>
#include <cstdint>
#include <cstddef>

// Problem constants (match reference)
#define BB   8
#define NN   256
#define DD   512
#define HH   64
#define DHD  8
#define LL   15
#define FFD  2048
#define KK   128

typedef __attribute__((ext_vector_type(16))) __bf16 v16bf;
typedef __attribute__((ext_vector_type(8)))  __bf16 v8bf;
typedef __attribute__((ext_vector_type(8)))  float  v8f;

__device__ inline v8f zero_v8f() {
  v8f v;
#pragma unroll
  for (int i = 0; i < 8; ++i) v[i] = 0.0f;
  return v;
}
__device__ inline v8bf zero_v8bf() {
  v8bf v;
#pragma unroll
  for (int i = 0; i < 8; ++i) v[i] = (__bf16)0.0f;
  return v;
}
__device__ inline v16bf concat8(v8bf lo, v8bf hi) {
  return __builtin_shufflevector(lo, hi, 0, 1, 2, 3, 4, 5, 6, 7,
                                 8, 9, 10, 11, 12, 13, 14, 15);
}

// CDNA5 async VMEM->LDS copy (ASYNCcnt), 16 bytes per lane.
// LDS byte address = low 32 bits of the generic pointer (LDS aperture layout).
__device__ inline void async_copy_b128(const void* gptr, void* lptr) {
  uint32_t la = (uint32_t)(uintptr_t)lptr;
  uint64_t ga = (uint64_t)(uintptr_t)gptr;
  asm volatile("global_load_async_to_lds_b128 %0, %1, off"
               :: "v"(la), "v"(ga)
               : "memory");
}
__device__ inline void wait_async0() {
  asm volatile("s_wait_asynccnt 0x0" ::: "memory");
}

// A-matrix 16x32 bf16 fragment (ISA 7.12.2). Row-major source, lda elements.
__device__ inline v16bf load_a_frag(const __bf16* A, int lda, int lane) {
  int g = lane >> 4, row = lane & 15;
  v8bf lo = *(const v8bf*)(A + row * lda + g * 8);
  v8bf hi = *(const v8bf*)(A + row * lda + 16 + g * 8);
  return concat8(lo, hi);
}
// B-matrix 32x16 bf16 fragment from a TRANSPOSED source BT[col][K].
__device__ inline v16bf load_b_frag_T(const __bf16* BT, int ldb, int lane) {
  int g = lane >> 4, col = lane & 15;
  v8bf lo = *(const v8bf*)(BT + col * ldb + g * 8);
  v8bf hi = *(const v8bf*)(BT + col * ldb + 16 + g * 8);
  return concat8(lo, hi);
}

__device__ inline float gelu_exact(float x) {
  return 0.5f * x * (1.0f + erff(x * 0.70710678118f));
}

// ------------------------------------------------- transpose + f32->bf16 convert
__global__ __launch_bounds__(256) void k_cvt_T(const float* __restrict__ src,
                                               __bf16* __restrict__ dst,
                                               int Kd, int Nd) {
  __shared__ float tile[32][33];
  int zt = blockIdx.z;
  const float* s = src + (size_t)zt * Kd * Nd;
  __bf16* d = dst + (size_t)zt * Kd * Nd;
  int n0 = blockIdx.x * 32, k0 = blockIdx.y * 32;
  int tx = threadIdx.x & 31, ty = threadIdx.x >> 5;   // 32 x 8
#pragma unroll
  for (int j = 0; j < 32; j += 8)
    tile[ty + j][tx] = s[(size_t)(k0 + ty + j) * Nd + n0 + tx];
  __syncthreads();
#pragma unroll
  for (int j = 0; j < 32; j += 8)
    d[(size_t)(n0 + ty + j) * Kd + k0 + tx] = (__bf16)tile[tx][ty + j];
}

// ------------------------------------------------- embedding gather
__global__ __launch_bounds__(256) void k_embed(const int* __restrict__ tok,
                                               const float* __restrict__ emb,
                                               float* __restrict__ out) {
  int i = blockIdx.x * 256 + threadIdx.x;   // over B*N*D
  int row = i / DD, c = i % DD;
  out[i] = emb[tok[row] * DD + c];
}

// ------------------------------------------------- layernorm (one row / block)
__global__ __launch_bounds__(256) void k_layernorm(const float* __restrict__ x,
                                                   const float* __restrict__ gam,
                                                   const float* __restrict__ bet,
                                                   float* __restrict__ outF,
                                                   __bf16* __restrict__ outB) {
  int row = blockIdx.x, tid = threadIdx.x;
  const float* xr = x + (size_t)row * DD;
  __shared__ float red[256];
  float v0 = xr[tid], v1 = xr[tid + 256];
  float s = v0 + v1, s2 = v0 * v0 + v1 * v1;
  red[tid] = s; __syncthreads();
  for (int o = 128; o > 0; o >>= 1) { if (tid < o) red[tid] += red[tid + o]; __syncthreads(); }
  float mean = red[0] * (1.0f / DD);
  __syncthreads();
  red[tid] = s2; __syncthreads();
  for (int o = 128; o > 0; o >>= 1) { if (tid < o) red[tid] += red[tid + o]; __syncthreads(); }
  float var = red[0] * (1.0f / DD) - mean * mean;
  float r = rsqrtf(var + 1e-5f);
#pragma unroll
  for (int j = 0; j < 2; ++j) {
    int c = tid + j * 256;
    float y = (xr[c] - mean) * r * gam[c] + bet[c];
    if (outF) outF[(size_t)row * DD + c] = y;
    if (outB) outB[(size_t)row * DD + c] = (__bf16)y;
  }
}

// ------------------------------------------------- generic bf16 WMMA GEMM
// C[M,Nd] = epi(A[M,Kd] @ W[Kd,Nd] + bias); W supplied TRANSPOSED (WT[Nd,Kd]).
// Tiles staged global->LDS with CDNA5 async-to-LDS, double-buffered so the
// DMA for slab k+1 overlaps the WMMAs on slab k.
#define GTM 128
#define GTN 64
__global__ __launch_bounds__(256) void k_gemm_bf16(const __bf16* __restrict__ A,
                                                   const __bf16* __restrict__ WT,
                                                   const float* __restrict__ bias,
                                                   const float* __restrict__ resid,
                                                   float* __restrict__ outF,
                                                   __bf16* __restrict__ outB,
                                                   int M, int Nd, int Kd, int do_gelu) {
  __shared__ __bf16 As[2][GTM][32];    // row-major A tiles (double buffered)
  __shared__ __bf16 BsT[2][GTN][32];   // transposed B tiles: [n][k]
  int tid = threadIdx.x, wv = tid >> 5, lane = tid & 31;
  int m0 = blockIdx.x * GTM, n0 = blockIdx.y * GTN;
  (void)M;
  v8f acc[4];
#pragma unroll
  for (int t = 0; t < 4; ++t) acc[t] = zero_v8f();

  // per-thread staging slots (8 bf16 = 16B each)
  int rowA0 = tid >> 2,          qa0 = (tid & 3) * 8;          // p = tid
  int rowA1 = (tid + 256) >> 2,  qa1 = qa0;                    // p = tid + 256
  int rowB  = tid >> 2,          qb  = qa0;                    // p = tid (< 256)

  auto issue = [&](int k0, int buf) {
    async_copy_b128(A  + (size_t)(m0 + rowA0) * Kd + k0 + qa0, &As[buf][rowA0][qa0]);
    async_copy_b128(A  + (size_t)(m0 + rowA1) * Kd + k0 + qa1, &As[buf][rowA1][qa1]);
    async_copy_b128(WT + (size_t)(n0 + rowB) * Kd + k0 + qb,  &BsT[buf][rowB][qb]);
  };

  issue(0, 0);
  int buf = 0;
  for (int k0 = 0; k0 < Kd; k0 += 32) {
    wait_async0();                       // slab for this iteration landed
    __syncthreads();                     // publish async LDS writes to all waves
    if (k0 + 32 < Kd) issue(k0 + 32, buf ^ 1);   // overlap DMA with WMMAs
    v16bf af = load_a_frag(&As[buf][wv * 16][0], 32, lane);
#pragma unroll
    for (int nt = 0; nt < 4; ++nt) {
      v16bf bfm = load_b_frag_T(&BsT[buf][nt * 16][0], 32, lane);
      acc[nt] = __builtin_amdgcn_wmma_f32_16x16x32_bf16(false, af, false, bfm,
                                                        (short)0, acc[nt], false, false);
    }
    __syncthreads();                     // all waves done reading buf
    buf ^= 1;
  }

  int g = lane >> 4, col = lane & 15;
#pragma unroll
  for (int nt = 0; nt < 4; ++nt) {
#pragma unroll
    for (int r = 0; r < 8; ++r) {
      int gr = m0 + wv * 16 + r + 8 * g;
      int gc = n0 + nt * 16 + col;
      float v = acc[nt][r] + (bias ? bias[gc] : 0.0f);
      if (do_gelu) v = gelu_exact(v);
      size_t gi = (size_t)gr * Nd + gc;
      if (resid) v += resid[gi];
      if (outF) outF[gi] = v;
      if (outB) outB[gi] = (__bf16)v;
    }
  }
}

// ------------------------------------------------- fused GBF -> proj1(gelu) -> proj2 -> bias0
__global__ __launch_bounds__(256) void k_gbf_bias(const float* __restrict__ dist,
                                                  const int* __restrict__ etype,
                                                  const float* __restrict__ gmul,
                                                  const float* __restrict__ gbias,
                                                  const float* __restrict__ means,
                                                  const float* __restrict__ stds,
                                                  const __bf16* __restrict__ W1T,
                                                  const float* __restrict__ b1,
                                                  const __bf16* __restrict__ W2T,
                                                  const float* __restrict__ b2,
                                                  float* __restrict__ am) {
  __shared__ __bf16 sW1T[KK * KK];
  __shared__ __bf16 sW2T[HH * KK];
  __shared__ __bf16 sG[16 * KK];
  __shared__ __bf16 sH[16 * KK];
  int tid = threadIdx.x, wv = tid >> 5, lane = tid & 31;

#pragma unroll
  for (int i = tid; i < KK * KK / 8; i += 256)    // 8 x b128 async per thread
    async_copy_b128((const uint4*)W1T + i, (uint4*)sW1T + i);
#pragma unroll
  for (int i = tid; i < HH * KK / 8; i += 256)    // 4 x b128 async per thread
    async_copy_b128((const uint4*)W2T + i, (uint4*)sW2T + i);

  long e0 = (long)blockIdx.x * 16;                // async DMA overlaps this math
#pragma unroll
  for (int i = tid; i < 16 * KK; i += 256) {
    int r = i >> 7, kk = i & (KK - 1);
    long e = e0 + r;
    float d = dist[e];
    int et = etype[e];
    float y = gmul[et] * d + gbias[et];
    float sd = fabsf(stds[kk]) + 1e-5f;
    float t = (y - means[kk]) / sd;
    float gv = expf(-0.5f * t * t) / (2.5066262f * sd);   // sqrt(2*3.14159)
    sG[r * KK + kk] = (__bf16)gv;
  }
  wait_async0();
  __syncthreads();

  // [16,128] = gelu(sG @ W1 + b1); one wave per 16-wide N tile
  v8f a1 = zero_v8f();
#pragma unroll
  for (int ks = 0; ks < 4; ++ks) {
    v16bf af  = load_a_frag(&sG[ks * 32], KK, lane);
    v16bf bfm = load_b_frag_T(&sW1T[(wv * 16) * KK + ks * 32], KK, lane);
    a1 = __builtin_amdgcn_wmma_f32_16x16x32_bf16(false, af, false, bfm,
                                                 (short)0, a1, false, false);
  }
  int g = lane >> 4, col = lane & 15;
#pragma unroll
  for (int r = 0; r < 8; ++r) {
    int row = r + 8 * g, n = wv * 16 + col;
    sH[row * KK + n] = (__bf16)gelu_exact(a1[r] + b1[n]);
  }
  __syncthreads();

  // [16,64] = sH @ W2 + b2 -> scatter (transposed) into am[b,h,i,j]
  if (wv < 4) {
    v8f a2 = zero_v8f();
#pragma unroll
    for (int ks = 0; ks < 4; ++ks) {
      v16bf af  = load_a_frag(&sH[ks * 32], KK, lane);
      v16bf bfm = load_b_frag_T(&sW2T[(wv * 16) * KK + ks * 32], KK, lane);
      a2 = __builtin_amdgcn_wmma_f32_16x16x32_bf16(false, af, false, bfm,
                                                   (short)0, a2, false, false);
    }
#pragma unroll
    for (int r = 0; r < 8; ++r) {
      int row = r + 8 * g, h = wv * 16 + col;
      long e = e0 + row;
      int b = (int)(e / ((long)NN * NN));
      int rem = (int)(e % ((long)NN * NN));
      int ii = rem / NN, jj = rem % NN;
      am[(((size_t)b * HH + h) * NN + ii) * NN + jj] = a2[r] + b2[h];
    }
  }
}

// ------------------------------------------------- Q/K repack [B,N,H,DH] f32 -> [B,H,N,DH] bf16
__global__ __launch_bounds__(256) void k_pack_qk(const float* __restrict__ q,
                                                 const float* __restrict__ k,
                                                 __bf16* __restrict__ qb,
                                                 __bf16* __restrict__ kb) {
  int i = blockIdx.x * 256 + threadIdx.x;   // over B*N*D
  int b = i / (NN * DD);
  int rem = i % (NN * DD);
  int n = rem / DD, c = rem % DD;
  int h = c / DHD, dh = c % DHD;
  size_t dst = (((size_t)b * HH + h) * NN + n) * DHD + dh;
  qb[dst] = (__bf16)(q[i] * 0.35355339059f);   // DH^-0.5
  kb[dst] = (__bf16)k[i];
}

// ------------------------------------------------- V repack -> vT [B,H,16,N] bf16 (dh rows 8..15 zero)
__global__ __launch_bounds__(256) void k_pack_vT(const float* __restrict__ v,
                                                 __bf16* __restrict__ vT) {
  int i = blockIdx.x * 256 + threadIdx.x;   // over B*H*16*N
  int n = i & (NN - 1);
  int rest = i >> 8;
  int dh = rest & 15; rest >>= 4;
  int h = rest & (HH - 1);
  int b = rest >> 6;
  float val = (dh < DHD)
      ? v[((size_t)(b * NN + n)) * DD + h * DHD + dh]
      : 0.0f;
  vT[i] = (__bf16)val;
}

// ------------------------------------------------- am += Q K^T  (one wave = one 16x16 tile)
__global__ __launch_bounds__(256) void k_attn_logits(const __bf16* __restrict__ qb,
                                                     const __bf16* __restrict__ kb,
                                                     float* __restrict__ am) {
  int wid = blockIdx.x * 8 + (threadIdx.x >> 5);
  int lane = threadIdx.x & 31;
  int tile = wid & 255, bh = wid >> 8;
  int qt = tile >> 4, kt = tile & 15;
  const __bf16* qp = qb + ((size_t)bh * NN + qt * 16) * DHD;
  const __bf16* kp = kb + ((size_t)bh * NN + kt * 16) * DHD;
  int g = lane >> 4, idx = lane & 15;
  v8bf z8 = zero_v8bf();
  v8bf aq  = (g == 0) ? *(const v8bf*)(qp + idx * DHD) : z8;
  v8bf bk_ = (g == 0) ? *(const v8bf*)(kp + idx * DHD) : z8;
  v16bf af  = concat8(aq, z8);
  v16bf bfm = concat8(bk_, z8);
  v8f acc = zero_v8f();
  acc = __builtin_amdgcn_wmma_f32_16x16x32_bf16(false, af, false, bfm,
                                                (short)0, acc, false, false);
  float* amp = am + ((size_t)bh * NN + qt * 16) * NN + kt * 16;
#pragma unroll
  for (int r = 0; r < 8; ++r) {
    int row = r + 8 * g;
    amp[(size_t)row * NN + idx] += acc[r];
  }
}

// ------------------------------------------------- row softmax over am -> probs (bf16)
__global__ __launch_bounds__(256) void k_softmax(const float* __restrict__ am,
                                                 __bf16* __restrict__ probs) {
  size_t row = blockIdx.x;
  int tid = threadIdx.x;
  __shared__ float red[256];
  float v = am[row * NN + tid];
  red[tid] = v; __syncthreads();
  for (int o = 128; o > 0; o >>= 1) { if (tid < o) red[tid] = fmaxf(red[tid], red[tid + o]); __syncthreads(); }
  float mx = red[0]; __syncthreads();
  float e = expf(v - mx);
  red[tid] = e; __syncthreads();
  for (int o = 128; o > 0; o >>= 1) { if (tid < o) red[tid] += red[tid + o]; __syncthreads(); }
  probs[row * NN + tid] = (__bf16)(e / red[0]);
}

// ------------------------------------------------- O = probs @ V  (one wave = 16 q-rows)
__global__ __launch_bounds__(256) void k_attn_av(const __bf16* __restrict__ probs,
                                                 const __bf16* __restrict__ vT,
                                                 __bf16* __restrict__ o) {
  int wid = blockIdx.x * 8 + (threadIdx.x >> 5);
  int lane = threadIdx.x & 31;
  int qt = wid & 15, bh = wid >> 4;
  int b = bh >> 6, h = bh & 63;
  const __bf16* pp = probs + ((size_t)bh * NN + qt * 16) * NN;
  const __bf16* vp = vT + (size_t)bh * 16 * NN;
  int g = lane >> 4, idx = lane & 15;
  v8f acc = zero_v8f();
#pragma unroll
  for (int ks = 0; ks < NN / 32; ++ks) {
    v16bf af  = load_a_frag(pp + ks * 32, NN, lane);
    v16bf bfm = load_b_frag_T(vp + ks * 32, NN, lane);
    acc = __builtin_amdgcn_wmma_f32_16x16x32_bf16(false, af, false, bfm,
                                                  (short)0, acc, false, false);
  }
#pragma unroll
  for (int r = 0; r < 8; ++r) {
    int row = r + 8 * g;
    if (idx < DHD)
      o[(((size_t)b * NN + qt * 16 + row) * HH + h) * DHD + idx] = (__bf16)acc[r];
  }
}

// ------------------------------------------------- pair output: [B,H,N,N] -> [B,N,N,H], -inf -> 0
__global__ __launch_bounds__(256) void k_pair_out(const float* __restrict__ am,
                                                  float* __restrict__ out) {
  size_t i = (size_t)blockIdx.x * 256 + threadIdx.x;   // output index [b,i,j,h]
  size_t h = i & (HH - 1);
  size_t rest = i >> 6;
  size_t j = rest % NN; rest /= NN;
  size_t ii = rest % NN;
  size_t b = rest / NN;
  float v = am[((b * HH + h) * NN + ii) * NN + j];
  if (isinf(v) && v < 0.0f) v = 0.0f;
  out[i] = v;
}

// ================================================================ host
extern "C" void kernel_launch(void* const* d_in, const int* in_sizes, int n_in,
                              void* d_out, int out_size, void* d_ws, size_t ws_size,
                              hipStream_t stream) {
  (void)in_sizes; (void)n_in; (void)out_size; (void)ws_size;

  const int*   src_tokens   = (const int*)d_in[0];
  const float* src_distance = (const float*)d_in[1];
  /* d_in[2] = src_coord: unused in infer mode */
  const int*   src_edge     = (const int*)d_in[3];
  const float* embed    = (const float*)d_in[4];
  const float* emb_ln_g = (const float*)d_in[5];
  const float* emb_ln_b = (const float*)d_in[6];
  const float* fin_ln_g = (const float*)d_in[7];
  const float* fin_ln_b = (const float*)d_in[8];
  const float* Wq = (const float*)d_in[9];   const float* bq = (const float*)d_in[10];
  const float* Wk = (const float*)d_in[11];  const float* bk = (const float*)d_in[12];
  const float* Wv = (const float*)d_in[13];  const float* bv = (const float*)d_in[14];
  const float* Wo = (const float*)d_in[15];  const float* bo = (const float*)d_in[16];
  const float* ln1_g = (const float*)d_in[17]; const float* ln1_b = (const float*)d_in[18];
  const float* ln2_g = (const float*)d_in[19]; const float* ln2_b = (const float*)d_in[20];
  const float* fc1_W = (const float*)d_in[21]; const float* fc1_b = (const float*)d_in[22];
  const float* fc2_W = (const float*)d_in[23]; const float* fc2_b = (const float*)d_in[24];
  const float* gbf_means = (const float*)d_in[25];
  const float* gbf_stds  = (const float*)d_in[26];
  const float* gbf_mul   = (const float*)d_in[27];
  const float* gbf_bias  = (const float*)d_in[28];
  const float* proj_W1 = (const float*)d_in[29]; const float* proj_b1 = (const float*)d_in[30];
  const float* proj_W2 = (const float*)d_in[31]; const float* proj_b2 = (const float*)d_in[32];

  const size_t TOK = (size_t)BB * NN;            // 2048 rows
  const size_t XE  = TOK * DD;                   // 1,048,576
  const size_t EE  = (size_t)BB * NN * NN;       // 524,288 edges
  const size_t AME = (size_t)BB * HH * NN * NN;  // 33,554,432

  char* wsb = (char*)d_ws;
  size_t off = 0;
  auto alloc = [&](size_t bytes) -> void* {
    void* p = wsb + off;
    off += bytes;
    off = (off + 255) & ~(size_t)255;
    return p;
  };

  // Transposed bf16 weights
  __bf16* wqT = (__bf16*)alloc((size_t)LL * DD * DD * 2);
  __bf16* wkT = (__bf16*)alloc((size_t)LL * DD * DD * 2);
  __bf16* wvT = (__bf16*)alloc((size_t)LL * DD * DD * 2);
  __bf16* woT = (__bf16*)alloc((size_t)LL * DD * DD * 2);
  __bf16* f1T = (__bf16*)alloc((size_t)LL * DD * FFD * 2);
  __bf16* f2T = (__bf16*)alloc((size_t)LL * FFD * DD * 2);
  __bf16* p1T = (__bf16*)alloc((size_t)KK * KK * 2);
  __bf16* p2T = (__bf16*)alloc((size_t)KK * HH * 2);
  float*  x   = (float*)alloc(XE * 4);
  float*  xt  = (float*)alloc(XE * 4);
  __bf16* hB  = (__bf16*)alloc(XE * 2);
  float*  qf  = (float*)alloc(XE * 4);
  float*  kf  = (float*)alloc(XE * 4);
  float*  vf  = (float*)alloc(XE * 4);
  __bf16* qB  = (__bf16*)alloc(XE * 2);
  __bf16* kB2 = (__bf16*)alloc(XE * 2);
  __bf16* vT  = (__bf16*)alloc((size_t)BB * HH * 16 * NN * 2);
  __bf16* oB  = (__bf16*)alloc(XE * 2);
  float*  am  = (float*)alloc(AME * 4);
  __bf16* prb = (__bf16*)alloc(AME * 2);
  __bf16* ffn = (__bf16*)alloc(TOK * FFD * 2);

  const int T = 256;
  // transpose-convert weights: src [z][Kd][Nd] -> dst [z][Nd][Kd]
  k_cvt_T<<<dim3(DD / 32, DD / 32, LL), T, 0, stream>>>(Wq, wqT, DD, DD);
  k_cvt_T<<<dim3(DD / 32, DD / 32, LL), T, 0, stream>>>(Wk, wkT, DD, DD);
  k_cvt_T<<<dim3(DD / 32, DD / 32, LL), T, 0, stream>>>(Wv, wvT, DD, DD);
  k_cvt_T<<<dim3(DD / 32, DD / 32, LL), T, 0, stream>>>(Wo, woT, DD, DD);
  k_cvt_T<<<dim3(FFD / 32, DD / 32, LL), T, 0, stream>>>(fc1_W, f1T, DD, FFD);
  k_cvt_T<<<dim3(DD / 32, FFD / 32, LL), T, 0, stream>>>(fc2_W, f2T, FFD, DD);
  k_cvt_T<<<dim3(KK / 32, KK / 32, 1), T, 0, stream>>>(proj_W1, p1T, KK, KK);
  k_cvt_T<<<dim3(HH / 32, KK / 32, 1), T, 0, stream>>>(proj_W2, p2T, KK, HH);

  k_embed<<<(unsigned)(XE / 256), T, 0, stream>>>(src_tokens, embed, xt);
  k_layernorm<<<(unsigned)TOK, T, 0, stream>>>(xt, emb_ln_g, emb_ln_b, x, nullptr);

  k_gbf_bias<<<(unsigned)(EE / 16), T, 0, stream>>>(src_distance, src_edge,
      gbf_mul, gbf_bias, gbf_means, gbf_stds, p1T, proj_b1, p2T, proj_b2, am);

  dim3 gD((unsigned)(TOK / GTM), DD / GTN);
  dim3 gF1((unsigned)(TOK / GTM), FFD / GTN);
  for (int l = 0; l < LL; ++l) {
    k_layernorm<<<(unsigned)TOK, T, 0, stream>>>(x, ln1_g + l * DD, ln1_b + l * DD,
                                                 nullptr, hB);
    k_gemm_bf16<<<gD, T, 0, stream>>>(hB, wqT + (size_t)l * DD * DD, bq + l * DD,
                                      nullptr, qf, nullptr, (int)TOK, DD, DD, 0);
    k_gemm_bf16<<<gD, T, 0, stream>>>(hB, wkT + (size_t)l * DD * DD, bk + l * DD,
                                      nullptr, kf, nullptr, (int)TOK, DD, DD, 0);
    k_gemm_bf16<<<gD, T, 0, stream>>>(hB, wvT + (size_t)l * DD * DD, bv + l * DD,
                                      nullptr, vf, nullptr, (int)TOK, DD, DD, 0);
    k_pack_qk<<<(unsigned)(XE / 256), T, 0, stream>>>(qf, kf, qB, kB2);
    k_pack_vT<<<(unsigned)((size_t)BB * HH * 16 * NN / 256), T, 0, stream>>>(vf, vT);

    k_attn_logits<<<(BB * HH * 16 * 16) / 8, T, 0, stream>>>(qB, kB2, am);
    k_softmax<<<BB * HH * NN, T, 0, stream>>>(am, prb);
    k_attn_av<<<(BB * HH * 16) / 8, T, 0, stream>>>(prb, vT, oB);

    k_gemm_bf16<<<gD, T, 0, stream>>>(oB, woT + (size_t)l * DD * DD, bo + l * DD,
                                      x, x, nullptr, (int)TOK, DD, DD, 0);
    k_layernorm<<<(unsigned)TOK, T, 0, stream>>>(x, ln2_g + l * DD, ln2_b + l * DD,
                                                 nullptr, hB);
    k_gemm_bf16<<<gF1, T, 0, stream>>>(hB, f1T + (size_t)l * DD * FFD, fc1_b + l * FFD,
                                       nullptr, nullptr, ffn, (int)TOK, FFD, DD, 1);
    k_gemm_bf16<<<gD, T, 0, stream>>>(ffn, f2T + (size_t)l * FFD * DD, fc2_b + l * DD,
                                      x, x, nullptr, (int)TOK, DD, FFD, 0);
  }

  k_layernorm<<<(unsigned)TOK, T, 0, stream>>>(x, fin_ln_g, fin_ln_b,
                                               (float*)d_out, nullptr);
  k_pair_out<<<(unsigned)(AME / 256), T, 0, stream>>>(am, (float*)d_out + XE);
}
